// SpatialAttention_13503377179057
// MI455X (gfx1250) — matrix operand order repaired and checked by reading, not compiled, not run on hardware
//
#include <hip/hip_runtime.h>
#include <cmath>

typedef __attribute__((ext_vector_type(16))) _Float16 v16h;
typedef __attribute__((ext_vector_type(8)))  _Float16 v8h;
typedef __attribute__((ext_vector_type(8)))  float    v8f;
typedef __attribute__((__vector_size__(16))) int      v4i;
typedef __attribute__((address_space(1)))    v4i      g_v4i;  // global
typedef __attribute__((address_space(3)))    v4i      l_v4i;  // LDS

static constexpr int BB = 4;
static constexpr int NN = 4096;
static constexpr int CC = 128;

#define WMMA_F16(A, B, C) \
  __builtin_amdgcn_wmma_f32_16x16x32_f16(false, (A), false, (B), (short)0, (C), false, false)

#if defined(__gfx1250__) && __has_builtin(__builtin_amdgcn_global_load_async_to_lds_b128)
#define HAVE_ASYNC_LDS 1
#else
#define HAVE_ASYNC_LDS 0
#endif

// ---------------------------------------------------------------------------
// Operand loaders for v_wmma_f32_16x16x32_f16 (wave32 layouts per CDNA5 ISA).
//
// A (16x32 f16): lane L holds row M = L&15; with kh = L>>4 the 16 elements
// map to K = kh*8 + i (i<8) and K = 16 + kh*8 + (i-8)  -> two contiguous
// 16-byte chunks of a row-major source.
//
// B (32x16 f16): lane L holds column N = L&15; with kh = L>>4 element i maps to
// K = kh*8 + i (i<8) and K = 16 + kh*8 + (i-8) -> two 16-byte chunks of row N
// of the *transposed* (N-major) source.
// ---------------------------------------------------------------------------
static __device__ inline v16h load_A_f16(const _Float16* base, int ld, int lane) {
  const int row = lane & 15, kh = (lane >> 4) & 1;
  const _Float16* p = base + (size_t)row * ld;
  v8h lo = *(const v8h*)(p + 8 * kh);
  v8h hi = *(const v8h*)(p + 16 + 8 * kh);
  v16h a;
#pragma unroll
  for (int i = 0; i < 8; ++i) { a[i] = lo[i]; a[8 + i] = hi[i]; }
  return a;
}

static __device__ inline v16h load_A_f32cvt(const float* base, int ld, int lane) {
  const int row = lane & 15, kh = (lane >> 4) & 1;
  const float* p = base + (size_t)row * ld;
  v16h a;
#pragma unroll
  for (int i = 0; i < 8; ++i) {
    a[i]     = (_Float16)p[8 * kh + i];
    a[8 + i] = (_Float16)p[16 + 8 * kh + i];
  }
  return a;
}

static __device__ inline v16h load_B_f16(const _Float16* baseT, int ld, int lane) {
  const int n = lane & 15, kh = (lane >> 4) & 1;
  const _Float16* p = baseT + (size_t)n * ld + 8 * kh;
  v8h lo = *(const v8h*)(p);
  v8h hi = *(const v8h*)(p + 16);
  v16h b;
#pragma unroll
  for (int i = 0; i < 8; ++i) { b[i] = lo[i]; b[8 + i] = hi[i]; }
  return b;
}

static __device__ inline v16h load_B_f32cvt(const float* baseT, int ld, int lane) {
  const int n = lane & 15, kh = (lane >> 4) & 1;
  const float* p = baseT + (size_t)n * ld + 8 * kh;
  v16h b;
#pragma unroll
  for (int i = 0; i < 8; ++i) { b[i] = (_Float16)p[i]; b[8 + i] = (_Float16)p[16 + i]; }
  return b;
}

// ---------------------------------------------------------------------------
// Kernel 1: q/k/v = GELU_exact(x @ W^T + b), stored as f16.
// One wave -> one 16-row x 128-col output tile of one of {q,k,v}.
// ---------------------------------------------------------------------------
__global__ __launch_bounds__(256) void qkv_proj_kernel(
    const float* __restrict__ x,
    const float* __restrict__ Wq, const float* __restrict__ bq,
    const float* __restrict__ Wk, const float* __restrict__ bk,
    const float* __restrict__ Wv, const float* __restrict__ bv,
    _Float16* __restrict__ q, _Float16* __restrict__ k, _Float16* __restrict__ v) {
  const int lane = threadIdx.x & 31;
  const int w    = blockIdx.x * (blockDim.x >> 5) + (threadIdx.x >> 5);
  const int m     = w >> 10;    // 0..2 : which projection
  const int strip = w & 1023;   // 16-row strip of the 16384 x-rows

  const float* W; const float* bias; _Float16* out;
  if (m == 0)      { W = Wq; bias = bq; out = q; }
  else if (m == 1) { W = Wk; bias = bk; out = k; }
  else             { W = Wv; bias = bv; out = v; }

  const float* xrow = x + (size_t)strip * 16 * CC;
  v16h A[4];
#pragma unroll
  for (int kk = 0; kk < 4; ++kk) A[kk] = load_A_f32cvt(xrow + kk * 32, CC, lane);

  const int nn = lane & 15, hi = (lane >> 4) & 1;
#pragma unroll
  for (int nt = 0; nt < 8; ++nt) {
    v8f acc = {};
#pragma unroll
    for (int kk = 0; kk < 4; ++kk)
      acc = WMMA_F16(A[kk], load_B_f32cvt(W + (size_t)(nt * 16) * CC + kk * 32, CC, lane), acc);
    const float bcol = bias[nt * 16 + nn];
#pragma unroll
    for (int r = 0; r < 8; ++r) {
      float t = acc[r] + bcol;
      float g = 0.5f * t * (1.0f + erff(t * 0.70710678118654752f));  // exact GELU
      const int row = strip * 16 + r + 8 * hi;
      out[(size_t)row * CC + nt * 16 + nn] = (_Float16)g;
    }
  }
}

// ---------------------------------------------------------------------------
// Kernel 2: flash attention, out = softmax(q k^T) v   (unscaled).
// Block = 8 waves x 32 lanes; each wave owns 16 query rows (128 rows/block);
// block iterates over 64-key tiles. K async-copied row-major into LDS
// (perfect B operand for S = QK^T); V staged transposed (perfect B operand
// for PV); P bounced through per-wave LDS to re-layout C/D -> A operand.
// ---------------------------------------------------------------------------
__global__ __launch_bounds__(256) void flash_attn_kernel(
    const _Float16* __restrict__ q, const _Float16* __restrict__ k,
    const _Float16* __restrict__ v, float* __restrict__ out) {
  __shared__ _Float16 Ks[64 * CC];      // [key][d]  16 KB
  __shared__ _Float16 Vt[CC][72];       // [d][key]  (pad 8) 18 KB
  __shared__ _Float16 Ps[8][16 * 64];   // per-wave P tile   16 KB

  const int tid  = threadIdx.x;
  const int lane = tid & 31;
  const int wave = tid >> 5;
  const int nn   = lane & 15, hi = (lane >> 4) & 1;
  const int batch = blockIdx.y;
  const int rowbase = blockIdx.x * 128 + wave * 16;

  const size_t batch_off = (size_t)batch * NN * CC;

  // Q tile in registers (16 rows x 128 d, as four A operands)
  v16h Aq[4];
#pragma unroll
  for (int kk = 0; kk < 4; ++kk)
    Aq[kk] = load_A_f16(q + batch_off + (size_t)rowbase * CC + kk * 32, CC, lane);

  v8f Oacc[8];
#pragma unroll
  for (int d = 0; d < 8; ++d) Oacc[d] = (v8f){};
  float m_i[8], l_i[8];
#pragma unroll
  for (int r = 0; r < 8; ++r) { m_i[r] = -1.0e30f; l_i[r] = 0.0f; }

  for (int kt = 0; kt < NN / 64; ++kt) {
    __syncthreads();  // protect previous tile readers (WAR)

    // ---- stage K tile row-major (16 KB): CDNA5 async copy if available
#if HAVE_ASYNC_LDS
    {
      const char* gsrc = (const char*)(k + batch_off + (size_t)kt * 64 * CC) + tid * 16;
      char* lds = (char*)&Ks[0] + tid * 16;
#pragma unroll
      for (int i = 0; i < 4; ++i)
        __builtin_amdgcn_global_load_async_to_lds_b128(
            (g_v4i*)(gsrc + i * 4096),
            (l_v4i*)(lds + i * 4096),
            0, 0);
      asm volatile("s_wait_asynccnt 0" ::: "memory");
    }
#else
    {
      const uint4* src = (const uint4*)(k + batch_off + (size_t)kt * 64 * CC);
      uint4* dst = (uint4*)Ks;
#pragma unroll
      for (int i = 0; i < 4; ++i) dst[tid + i * 256] = src[tid + i * 256];
    }
#endif
    // ---- stage V tile transposed: Vt[d][key]
    {
      const _Float16* vsrc = v + batch_off + (size_t)kt * 64 * CC;
#pragma unroll
      for (int i = 0; i < 4; ++i) {
        const int chunk = tid + i * 256;          // 0..1023
        const int key = chunk >> 4;               // 0..63
        const int db  = (chunk & 15) * 8;         // 0..120
        v8h vals = *(const v8h*)(vsrc + (size_t)key * CC + db);
#pragma unroll
        for (int j = 0; j < 8; ++j) Vt[db + j][key] = vals[j];
      }
    }
    __syncthreads();

    // ---- S = Q K^T for this wave's 16 rows vs 64 keys (4 C/D frags)
    v8f sfrag[4];
#pragma unroll
    for (int nk = 0; nk < 4; ++nk) {
      v8f s = {};
#pragma unroll
      for (int kk = 0; kk < 4; ++kk)
        s = WMMA_F16(Aq[kk], load_B_f16(Ks + (size_t)(nk * 16) * CC + kk * 32, CC, lane), s);
      sfrag[nk] = s;
    }

    // ---- online softmax: row = (r + 8*hi) spans the 16 lanes of a half-wave
    float mrow[8];
#pragma unroll
    for (int r = 0; r < 8; ++r)
      mrow[r] = fmaxf(fmaxf(sfrag[0][r], sfrag[1][r]), fmaxf(sfrag[2][r], sfrag[3][r]));
#pragma unroll
    for (int off = 1; off < 16; off <<= 1)
#pragma unroll
      for (int r = 0; r < 8; ++r)
        mrow[r] = fmaxf(mrow[r], __shfl_xor(mrow[r], off, 32));

    float rsum[8];
#pragma unroll
    for (int r = 0; r < 8; ++r) {
      const float mnew  = fmaxf(m_i[r], mrow[r]);
      const float scale = __expf(m_i[r] - mnew);
      m_i[r] = mnew;
      float rs = 0.0f;
#pragma unroll
      for (int nk = 0; nk < 4; ++nk) {
        const float p = __expf(sfrag[nk][r] - mnew);
        sfrag[nk][r] = p;
        rs += p;
      }
      rsum[r] = rs;
      l_i[r] *= scale;
#pragma unroll
      for (int d = 0; d < 8; ++d) Oacc[d][r] *= scale;
    }
#pragma unroll
    for (int off = 1; off < 16; off <<= 1)
#pragma unroll
      for (int r = 0; r < 8; ++r)
        rsum[r] += __shfl_xor(rsum[r], off, 32);
#pragma unroll
    for (int r = 0; r < 8; ++r) l_i[r] += rsum[r];

    // ---- bounce P through LDS: C/D layout -> row-major [16][64]
    _Float16* pbuf = &Ps[wave][0];
#pragma unroll
    for (int nk = 0; nk < 4; ++nk)
#pragma unroll
      for (int r = 0; r < 8; ++r)
        pbuf[(r + 8 * hi) * 64 + nk * 16 + nn] = (_Float16)sfrag[nk][r];
    asm volatile("s_wait_dscnt 0" ::: "memory");  // same-wave LDS RAW

    const v16h Ap0 = load_A_f16(pbuf, 64, lane);       // keys  0..31
    const v16h Ap1 = load_A_f16(pbuf + 32, 64, lane);  // keys 32..63

    // ---- O += P V  (Vt rows are perfect B operands)
#pragma unroll
    for (int d = 0; d < 8; ++d) {
      Oacc[d] = WMMA_F16(Ap0, load_B_f16(&Vt[d * 16][0], 72, lane), Oacc[d]);
      Oacc[d] = WMMA_F16(Ap1, load_B_f16(&Vt[d * 16][0] + 32, 72, lane), Oacc[d]);
    }
  }

  // ---- finalize: divide by softmax denominator, store fp32
#pragma unroll
  for (int d = 0; d < 8; ++d) {
#pragma unroll
    for (int r = 0; r < 8; ++r) {
      const int row = rowbase + r + 8 * hi;
      out[batch_off + (size_t)row * CC + d * 16 + nn] = Oacc[d][r] / l_i[r];
    }
  }
}

// ---------------------------------------------------------------------------
extern "C" void kernel_launch(void* const* d_in, const int* in_sizes, int n_in,
                              void* d_out, int out_size, void* d_ws, size_t ws_size,
                              hipStream_t stream) {
  const float* x  = (const float*)d_in[0];
  const float* Wq = (const float*)d_in[1];
  const float* bq = (const float*)d_in[2];
  const float* Wk = (const float*)d_in[3];
  const float* bk = (const float*)d_in[4];
  const float* Wv = (const float*)d_in[5];
  const float* bv = (const float*)d_in[6];

  const size_t BNC = (size_t)BB * NN * CC;  // 2,097,152 elements
  _Float16* q = (_Float16*)d_ws;
  _Float16* k = q + BNC;
  _Float16* v = k + BNC;

  // 3 * (16384/16) = 3072 waves, 8 waves per 256-thread block -> 384 blocks
  qkv_proj_kernel<<<384, 256, 0, stream>>>(x, Wq, bq, Wk, bk, Wv, bv, q, k, v);

  // 128 query rows per block (8 waves x 16), 32 row-tiles x 4 batches
  flash_attn_kernel<<<dim3(NN / 128, BB), 256, 0, stream>>>(q, k, v, (float*)d_out);
}
